// MLA_21973052686769
// MI455X (gfx1250) — matrix operand order repaired and checked by reading, not compiled, not run on hardware
//
#include <hip/hip_runtime.h>
#include <hip/hip_bf16.h>
#include <cstdint>

// ---------- types ----------
typedef __attribute__((ext_vector_type(8)))  float        v8f;
typedef __attribute__((ext_vector_type(16))) __bf16       v16bf;
typedef __attribute__((ext_vector_type(4)))  unsigned int uint4v;
typedef __attribute__((ext_vector_type(4)))  float        float4v;
typedef __attribute__((ext_vector_type(4)))  unsigned int u32x4;
typedef __attribute__((ext_vector_type(8)))  int          i32x8;
typedef __attribute__((ext_vector_type(4)))  int          i32x4;

union Frag { uint4v u[2]; v16bf v; };

// native f32 -> bf16 (backend emits gfx1250 bf16 cvt ops)
__device__ __forceinline__ __bf16 f2bf(float x) { return (__bf16)x; }

__device__ __forceinline__ v8f wmma_bf16(v16bf a, v16bf b, v8f c) {
  return __builtin_amdgcn_wmma_f32_16x16x32_bf16(false, a, false, b, (short)0, c,
                                                 false, false);
}

// ---------- Tensor Data Mover support ----------
#if defined(__has_builtin)
#if __has_builtin(__builtin_amdgcn_tensor_load_to_lds) && \
    __has_builtin(__builtin_amdgcn_s_wait_tensorcnt)
#define USE_TDM 1
#endif
#endif
#ifndef USE_TDM
#define USE_TDM 0
#endif

#if USE_TDM
// 2D tile load: tile_x contiguous f32 elements per line, tile_y lines,
// line stride = stride_elems. Optional LDS padding: +1 DWORD every 64 DWORDs.
__device__ __forceinline__ void tdm_load_2d(unsigned lds_off, const float* g,
                                            int tile_x, int tile_y,
                                            unsigned stride_elems, bool pad) {
  unsigned long long ga = (unsigned long long)(uintptr_t)g;
  u32x4 g0;
  g0[0] = 1u;                                             // count=1, user mode
  g0[1] = lds_off;                                        // lds_addr
  g0[2] = (unsigned)ga;                                   // global_addr[31:0]
  g0[3] = ((unsigned)(ga >> 32) & 0x1ffffffu) | (2u << 30);  // ga[56:32] | type=2
  i32x8 g1;
  unsigned w0 = (2u << 16);                               // data_size=4B
  if (pad) w0 |= (1u << 20) | (5u << 22);                 // pad_en, every 64 DW, +1 DW
  const unsigned td0 = 0x7fffffffu, td1 = 0x7fffffffu;    // huge dims: no OOB clip
  g1[0] = (int)w0;
  g1[1] = (int)((td0 & 0xffffu) << 16);                   // tensor_dim0 lo16
  g1[2] = (int)((td0 >> 16) | ((td1 & 0xffffu) << 16));   // td0 hi16 | td1 lo16
  g1[3] = (int)((td1 >> 16) | ((unsigned)tile_x << 16));  // td1 hi16 | tile_dim0
  g1[4] = (int)((unsigned)tile_y & 0xffffu);              // tile_dim1 (tile_dim2=0)
  g1[5] = (int)stride_elems;                              // dim0_stride lo32
  g1[6] = 0;
  g1[7] = 0;
  i32x4 z4 = {0, 0, 0, 0};
  i32x8 z8 = {0, 0, 0, 0, 0, 0, 0, 0};
  // 6-arg toolchain variant: (g0, g1, g2, g3, extra_group, cpol)
  __builtin_amdgcn_tensor_load_to_lds(g0, g1, z4, z4, z8, 0);
}
#endif

// =====================================================================
// GEMM: C[M,N] = A[M,K] x B[K,N], f32 in memory, bf16 WMMA, f32 accum.
// 64x64 block tile, K-step 32, 128 threads (4 waves), wave tile 32x32.
// TDM path: double-buffered async tile loads (TENSORcnt pipeline).
// =====================================================================
__global__ __launch_bounds__(128) void gemm_bf16(const float* __restrict__ A,
                                                 const float* __restrict__ B,
                                                 float* __restrict__ C,
                                                 int M, int N, int K) {
  __shared__ alignas(16) float Asf[2][64][32];      // row-major M x K
  __shared__ alignas(16) float Bsf[2][32 * 65];     // K x N, row stride 65 (padded)
  const int tid  = threadIdx.x;
  const int wave = tid >> 5;
  const int lane = tid & 31;
  const int hl   = lane >> 4;
  const int l16  = lane & 15;
  const int bm = blockIdx.y * 64, bn = blockIdx.x * 64;
  const int wr = (wave >> 1) * 32, wc = (wave & 1) * 32;
  const int kb8 = hl * 8, kb16 = hl * 16;
  const int T = K >> 5;

  const v8f vzero = {0.f, 0.f, 0.f, 0.f, 0.f, 0.f, 0.f, 0.f};
  v8f acc[2][2];
#pragma unroll
  for (int i = 0; i < 2; ++i)
#pragma unroll
    for (int j = 0; j < 2; ++j) acc[i][j] = vzero;

  int buf = 0;
#if USE_TDM
  if (wave == 0) {
    tdm_load_2d((unsigned)(uintptr_t)&Asf[0][0][0], A + (size_t)bm * K, 32, 64,
                (unsigned)K, false);
    tdm_load_2d((unsigned)(uintptr_t)&Bsf[0][0], B + bn, 64, 32, (unsigned)N, true);
  }
#endif

  for (int t = 0; t < T; ++t) {
#if USE_TDM
    if (wave == 0) {
      if (t + 1 < T) {
        tdm_load_2d((unsigned)(uintptr_t)&Asf[buf ^ 1][0][0],
                    A + (size_t)bm * K + (t + 1) * 32, 32, 64, (unsigned)K, false);
        tdm_load_2d((unsigned)(uintptr_t)&Bsf[buf ^ 1][0],
                    B + (size_t)(t + 1) * 32 * N + bn, 64, 32, (unsigned)N, true);
        __builtin_amdgcn_s_wait_tensorcnt(2);  // in-order: current tile done
      } else {
        __builtin_amdgcn_s_wait_tensorcnt(0);
      }
    }
    __syncthreads();
#else
    {
      const int k0 = t * 32;
#pragma unroll
      for (int it = 0; it < 4; ++it) {
        int i = tid + it * 128;
        int r = i >> 3, c = (i & 7) << 2;
        *(float4v*)&Asf[0][r][c] = *(const float4v*)(A + (size_t)(bm + r) * K + k0 + c);
      }
#pragma unroll
      for (int it = 0; it < 4; ++it) {
        int i = tid + it * 128;
        int kk = i >> 4, n4 = (i & 15) << 2;
        float4v f = *(const float4v*)(B + (size_t)(k0 + kk) * N + bn + n4);
        Bsf[0][kk * 65 + n4 + 0] = f[0];
        Bsf[0][kk * 65 + n4 + 1] = f[1];
        Bsf[0][kk * 65 + n4 + 2] = f[2];
        Bsf[0][kk * 65 + n4 + 3] = f[3];
      }
    }
    __syncthreads();
#endif

    // A-frag (row = lane%16, K chunks kb8 and 16+kb8), convert f32->bf16
    v16bf av[2], bv[2];
#pragma unroll
    for (int i = 0; i < 2; ++i) {
      const float* rp = &Asf[buf][wr + 16 * i + l16][0];
      float4v c0 = *(const float4v*)(rp + kb8);
      float4v c1 = *(const float4v*)(rp + kb8 + 4);
      float4v c2 = *(const float4v*)(rp + 16 + kb8);
      float4v c3 = *(const float4v*)(rp + 16 + kb8 + 4);
#pragma unroll
      for (int j = 0; j < 4; ++j) {
        av[i][j]      = f2bf(c0[j]);
        av[i][4 + j]  = f2bf(c1[j]);
        av[i][8 + j]  = f2bf(c2[j]);
        av[i][12 + j] = f2bf(c3[j]);
      }
    }
    // B-frag (col = lane%16, K rows kb8..kb8+7 and 16+kb8..16+kb8+7, padded stride 65 -> no bank conflicts)
#pragma unroll
    for (int j2 = 0; j2 < 2; ++j2) {
      const float* bp = &Bsf[buf][kb8 * 65 + wc + 16 * j2 + l16];
#pragma unroll
      for (int i = 0; i < 16; ++i) bv[j2][i] = f2bf(bp[((i & 7) + ((i >> 3) << 4)) * 65]);
    }
#pragma unroll
    for (int i = 0; i < 2; ++i)
#pragma unroll
      for (int j = 0; j < 2; ++j)
        acc[i][j] = wmma_bf16(av[i], bv[j], acc[i][j]);
    __syncthreads();
#if USE_TDM
    buf ^= 1;
#endif
  }

  // C layout: element r -> M = r + 8*(lane/16), N = lane%16
#pragma unroll
  for (int i = 0; i < 2; ++i)
#pragma unroll
    for (int j = 0; j < 2; ++j)
#pragma unroll
      for (int r = 0; r < 8; ++r)
        C[(size_t)(bm + wr + 16 * i + r + 8 * hl) * N + (bn + wc + 16 * j + l16)] =
            acc[i][j][r];
}

// =====================================================================
// RMSNorm over rows of 512 (in place)
// =====================================================================
__global__ __launch_bounds__(128) void rmsnorm_kernel(float* __restrict__ t,
                                                      const float* __restrict__ w) {
  __shared__ float red[128];
  float* p = t + (size_t)blockIdx.x * 512;
  float s = 0.f;
  for (int i = threadIdx.x; i < 512; i += 128) {
    float x = p[i];
    s += x * x;
  }
  red[threadIdx.x] = s;
  __syncthreads();
  for (int st = 64; st > 0; st >>= 1) {
    if (threadIdx.x < st) red[threadIdx.x] += red[threadIdx.x + st];
    __syncthreads();
  }
  float inv = rsqrtf(red[0] * (1.0f / 512.0f) + 1e-6f);
  for (int i = threadIdx.x; i < 512; i += 128) p[i] = p[i] * inv * w[i];
}

// =====================================================================
// RoPE (interleaved pairs) in place: t is [4096 rows][16 heads][128]
// =====================================================================
__global__ __launch_bounds__(256) void rope_kernel(float* __restrict__ t,
                                                   const float* __restrict__ fc,
                                                   const float* __restrict__ fs) {
  int idx = blockIdx.x * 256 + threadIdx.x;  // 4096*16*64 total
  int row = idx >> 10;
  int rem = idx & 1023;
  int h = rem >> 6;
  int i = rem & 63;
  int s = row & 2047;
  size_t base = ((size_t)row * 16 + h) * 128 + 2 * i;
  float x1 = t[base], x2 = t[base + 1];
  float c = fc[s * 64 + i], sn = fs[s * 64 + i];
  t[base]     = x1 * c - x2 * sn;
  t[base + 1] = x1 * sn + x2 * c;
}

// =====================================================================
// Flash attention, causal. One wave per 16 query rows per (b,h).
// q,k,v: [4096][16][128] f32 (q,k roped). o: [4096][2048] f32.
// =====================================================================
__global__ __launch_bounds__(32) void mla_attn(const float* __restrict__ q,
                                               const float* __restrict__ k,
                                               const float* __restrict__ v,
                                               float* __restrict__ o) {
  __shared__ alignas(16) __bf16 Plds[16][32];
  const int lane = threadIdx.x;
  const int hl = lane >> 4;
  const int l16 = lane & 15;
  const int qt = blockIdx.x;
  const int bh = blockIdx.y;
  const int b = bh >> 4, h = bh & 15;
  const int qbase = qt * 16;
  const float scale = 0.08838834764831845f;  // 1/sqrt(128)
  const int kb8 = hl * 8;
  const int kb16 = hl * 16;

  v16bf qa[4];
  const float* qp = q + ((size_t)(b * 2048 + qbase + l16) * 16 + h) * 128;
#pragma unroll
  for (int c = 0; c < 4; ++c)
#pragma unroll
    for (int j = 0; j < 8; ++j) {
      qa[c][j]     = f2bf(qp[c * 32 + kb16 + j]);
      qa[c][j + 8] = f2bf(qp[c * 32 + kb16 + 8 + j]);
    }

  const v8f vzero = {0.f, 0.f, 0.f, 0.f, 0.f, 0.f, 0.f, 0.f};
  float m[8], lsum[8];
  v8f oacc[8];
#pragma unroll
  for (int r = 0; r < 8; ++r) { m[r] = -3.0e38f; lsum[r] = 0.f; }
#pragma unroll
  for (int n = 0; n < 8; ++n) oacc[n] = vzero;

  const int nkb = (qbase + 16 + 31) >> 5;
  for (int blk = 0; blk < nkb; ++blk) {
    const int key0 = blk * 32;
    v8f s0 = vzero, s1 = vzero;
#pragma unroll
    for (int c = 0; c < 4; ++c) {
      v16bf kb0, kb1;
      const float* kp0 =
          k + ((size_t)(b * 2048 + key0 + l16) * 16 + h) * 128 + c * 32 + kb16;
      const float* kp1 = kp0 + (size_t)16 * 16 * 128;
#pragma unroll
      for (int i = 0; i < 16; ++i) {
        kb0[i] = f2bf(kp0[i]);
        kb1[i] = f2bf(kp1[i]);
      }
      s0 = wmma_bf16(qa[c], kb0, s0);
      s1 = wmma_bf16(qa[c], kb1, s1);
    }
#pragma unroll
    for (int r = 0; r < 8; ++r) {
      int qrow = qbase + r + 8 * hl;
      float a0 = (key0 + l16 <= qrow)      ? s0[r] * scale : -3.0e38f;
      float a1 = (key0 + 16 + l16 <= qrow) ? s1[r] * scale : -3.0e38f;
      float mb = fmaxf(a0, a1);
#pragma unroll
      for (int off = 8; off > 0; off >>= 1) mb = fmaxf(mb, __shfl_xor(mb, off, 32));
      float mn = fmaxf(m[r], mb);
      float sc = __expf(m[r] - mn);
      float p0 = __expf(a0 - mn);
      float p1 = __expf(a1 - mn);
      float ps = p0 + p1;
#pragma unroll
      for (int off = 8; off > 0; off >>= 1) ps += __shfl_xor(ps, off, 32);
      lsum[r] = lsum[r] * sc + ps;
      m[r] = mn;
#pragma unroll
      for (int n = 0; n < 8; ++n) oacc[n][r] *= sc;
      int rl = r + 8 * hl;
      Plds[rl][l16]      = f2bf(p0);
      Plds[rl][16 + l16] = f2bf(p1);
    }
    Frag pa;
    const uint4v* pr = (const uint4v*)(&Plds[l16][0]);
    pa.u[0] = pr[2 * hl];
    pa.u[1] = pr[2 * hl + 1];
#pragma unroll
    for (int n = 0; n < 8; ++n) {
      v16bf vb;
      const float* vp =
          v + ((size_t)(b * 2048 + key0 + kb16) * 16 + h) * 128 + n * 16 + l16;
#pragma unroll
      for (int i = 0; i < 16; ++i) vb[i] = f2bf(vp[(size_t)i * 2048]);
      oacc[n] = wmma_bf16(pa.v, vb, oacc[n]);
    }
  }

#pragma unroll
  for (int r = 0; r < 8; ++r) {
    float inv = 1.0f / lsum[r];
    size_t row = (size_t)(b * 2048 + qbase + r + 8 * hl);
#pragma unroll
    for (int n = 0; n < 8; ++n)
      o[row * 2048 + h * 128 + n * 16 + l16] = oacc[n][r] * inv;
  }
}

// =====================================================================
// launcher
// =====================================================================
extern "C" void kernel_launch(void* const* d_in, const int* in_sizes, int n_in,
                              void* d_out, int out_size, void* d_ws, size_t ws_size,
                              hipStream_t stream) {
  const float* x     = (const float*)d_in[0];
  const float* fc    = (const float*)d_in[1];
  const float* fs    = (const float*)d_in[2];
  const float* w_kvc = (const float*)d_in[3];
  const float* kvnw  = (const float*)d_in[4];
  const float* w_kup = (const float*)d_in[5];
  const float* w_vup = (const float*)d_in[6];
  const float* w_q   = (const float*)d_in[7];
  const float* w_out = (const float*)d_in[8];
  float* out = (float*)d_out;

  float* ws = (float*)d_ws;
  float* kv = ws;                            // 4096*512
  float* q  = kv + (size_t)4096 * 512;       // 4096*2048
  float* kk = q + (size_t)4096 * 2048;       // 4096*2048
  float* vv = kk + (size_t)4096 * 2048;      // 4096*2048
  float* at = vv + (size_t)4096 * 2048;      // 4096*2048

  gemm_bf16<<<dim3(512 / 64, 4096 / 64), dim3(128), 0, stream>>>(x, w_kvc, kv,
                                                                 4096, 512, 2048);
  rmsnorm_kernel<<<dim3(4096), dim3(128), 0, stream>>>(kv, kvnw);
  gemm_bf16<<<dim3(2048 / 64, 4096 / 64), dim3(128), 0, stream>>>(kv, w_kup, kk,
                                                                  4096, 2048, 512);
  gemm_bf16<<<dim3(2048 / 64, 4096 / 64), dim3(128), 0, stream>>>(kv, w_vup, vv,
                                                                  4096, 2048, 512);
  gemm_bf16<<<dim3(2048 / 64, 4096 / 64), dim3(128), 0, stream>>>(x, w_q, q,
                                                                  4096, 2048, 2048);
  rope_kernel<<<dim3(16384), dim3(256), 0, stream>>>(q, fc, fs);
  rope_kernel<<<dim3(16384), dim3(256), 0, stream>>>(kk, fc, fs);
  mla_attn<<<dim3(128, 32), dim3(32), 0, stream>>>(q, kk, vv, at);
  gemm_bf16<<<dim3(2048 / 64, 4096 / 64), dim3(128), 0, stream>>>(at, w_out, out,
                                                                  4096, 2048, 2048);
}
